// Downsample_PASA_group_softmax_46986942218656
// MI455X (gfx1250) — matrix operand order repaired and checked by reading, not compiled, not run on hardware
//
#include <hip/hip_runtime.h>
#include <hip/hip_bf16.h>
#include <math.h>
#include <stdint.h>

typedef __attribute__((ext_vector_type(16))) _Float16 v16h;
typedef __attribute__((ext_vector_type(8)))  float    v8f;

union Frag { v16h h; unsigned int u[8]; };

#define A_STRIDE 66   // 64 kk + pad (halfs); 33 dwords/row -> conflict-free frag reads
#define B_STRIDE 68   // 64 kk + pad (halfs); 136B/row, 8B-aligned for async b64 stores

// ---------------------------------------------------------------------------
// Kernel 0: pack conv weights (54,64,3,3,3) f32 -> f16 tiles Bp[p][n][c]
// Global K order is tap-major: k = p*64 + c, so one K64 tile == one tap.
// ---------------------------------------------------------------------------
__global__ __launch_bounds__(256) void pasa_pack_b(const float* __restrict__ w,
                                                   _Float16* __restrict__ Bp) {
    int idx = blockIdx.x * 256 + threadIdx.x;
    if (idx >= 27 * 64 * 64) return;
    int p = idx >> 12;           // tap 0..26
    int n = (idx >> 6) & 63;     // padded cout
    int c = idx & 63;            // cin
    float v = 0.0f;
    if (n < 54) v = w[(n * 64 + c) * 27 + p];
    Bp[idx] = (_Float16)v;
}

// ---------------------------------------------------------------------------
// Kernel 1: im2col-GEMM via v_wmma_f32_16x16x32_f16.
// Block: 128 threads (4 waves). Output tile M=64 (two w-rows: h, h+1),
// N=64, K=1728 walked tap-by-tap (K-step 64 = 1 tap x 64 cin).
// Wave (0..3) owns a 32x32 sub-tile: 8 WMMAs / iteration.
// B tile staged via global_load_async_to_lds_b64 (ASYNCcnt path).
// Also accumulates per-channel sum/sumsq for BN; sigma stored only at
// stride-2 voxels (d even, h even [wave<2 half], w even).
// ---------------------------------------------------------------------------
__global__ __launch_bounds__(128) void pasa_conv_wmma(
    const float* __restrict__ x, const _Float16* __restrict__ Bp,
    float* __restrict__ sigma, float* __restrict__ sumBuf,
    float* __restrict__ sumsqBuf) {

    __shared__ _Float16 Atile[64 * A_STRIDE];
    __shared__ _Float16 Btile[64 * B_STRIDE];
    __shared__ float chSum[64];
    __shared__ float chSumSq[64];

    const int tid  = threadIdx.x;
    const int lane = tid & 31;
    const int wave = tid >> 5;

    const int posB  = blockIdx.x * 64;     // 64 positions: (h,w=0..31),(h+1,w=0..31)
    const int nb    = posB >> 15;
    const int d     = (posB >> 10) & 31;
    const int hbase = (posB >> 5) & 31;    // even (block is 64-aligned)

    const int wld = tid & 31;              // loader: w coordinate
    const int ccq = tid >> 5;              // loader: channel sub-lane (0..3)

    if (tid < 64) { chSum[tid] = 0.0f; chSumSq[tid] = 0.0f; }

    // reflected w for tap offsets -1,0,+1 (reflect101)
    const int ww0 = (wld == 0) ? 1 : wld - 1;
    const int ww1 = wld;
    const int ww2 = (wld == 31) ? 30 : wld + 1;

    const float* xn = x + (size_t)nb * (64 * 32768);

    v8f acc00 = {}, acc01 = {}, acc10 = {}, acc11 = {};

    const int mBaseRow = (wave >> 1) * 32;   // wave M-half
    const int nBase    = (wave & 1) * 32;    // wave N-half
    const int mrow     = lane & 15;
    const int hs       = lane >> 4;

    for (int p = 0; p < 27; ++p) {
        const int di = p / 9;
        const int hi = (p - di * 9) / 3;
        const int wi = p - di * 9 - hi * 3;
        int dd  = d + di - 1;         dd  = dd  < 0 ? 1 : (dd  > 31 ? 30 : dd);
        int hh0 = hbase + hi - 1;     hh0 = hh0 < 0 ? 1 : hh0;              // <=31
        int hh1 = hbase + hi;         hh1 = hh1 > 31 ? 30 : hh1;            // >=0
        const int wsel = (wi == 0) ? ww0 : ((wi == 1) ? ww1 : ww2);

        __syncthreads();   // previous iteration's fragment reads complete

        // --- B tile: 64x64 f16 (one tap, contiguous 8KB) via async->LDS ---
        {
            const _Float16* bsrc = Bp + (size_t)p * 4096;
            #pragma unroll
            for (int i = 0; i < 8; ++i) {
                int j   = tid + i * 128;        // b64 chunk id (0..1023)
                int n   = j >> 4;               // 16 chunks per 64-half row
                int col = (j & 15) * 4;
                unsigned ldsaddr = (unsigned)(uintptr_t)(&Btile[n * B_STRIDE + col]);
                const _Float16* g = bsrc + (size_t)j * 4;
                asm volatile("global_load_async_to_lds_b64 %0, %1, off"
                             :: "v"(ldsaddr), "v"(g) : "memory");
            }
        }

        // --- A tile: 64 positions x 64 channels of this tap, f32->f16 ---
        {
            const float* s0 = xn + dd * 1024 + hh0 * 32 + wsel;
            const float* s1 = xn + dd * 1024 + hh1 * 32 + wsel;
            #pragma unroll
            for (int i = 0; i < 16; ++i) {
                int cc = ccq + 4 * i;
                float v0 = s0[(size_t)cc * 32768];
                float v1 = s1[(size_t)cc * 32768];
                Atile[wld * A_STRIDE + cc]        = (_Float16)v0;
                Atile[(32 + wld) * A_STRIDE + cc] = (_Float16)v1;
            }
        }

        asm volatile("s_wait_asynccnt 0" ::: "memory");
        __syncthreads();

        // --- 2 K-subtiles x (2M x 2N) = 8 WMMAs ---
        #pragma unroll
        for (int s = 0; s < 2; ++s) {
            Frag fa0, fa1, fb0, fb1;
            #pragma unroll
            for (int r = 0; r < 8; ++r) {
                // A 16x32 frag: VGPR r<4 -> K=2r, r>=4 -> K=2r+8; lanes16-31 +8
                int ka = s * 32 + ((r < 4) ? (2 * r) : (2 * r + 8)) + 8 * hs;
                fa0.u[r] = *(const unsigned int*)(&Atile[(mBaseRow + mrow) * A_STRIDE + ka]);
                fa1.u[r] = *(const unsigned int*)(&Atile[(mBaseRow + 16 + mrow) * A_STRIDE + ka]);
                // B 32x16 frag: VGPR r<4 -> K=2r, r>=4 -> K=2r+8; lanes16-31 +8
                int kb = s * 32 + ((r < 4) ? (2 * r) : (2 * r + 8)) + 8 * hs;
                fb0.u[r] = *(const unsigned int*)(&Btile[(nBase + mrow) * B_STRIDE + kb]);
                fb1.u[r] = *(const unsigned int*)(&Btile[(nBase + 16 + mrow) * B_STRIDE + kb]);
            }
            acc00 = __builtin_amdgcn_wmma_f32_16x16x32_f16(false, fa0.h, false, fb0.h,
                                                           (short)0, acc00, false, false);
            acc01 = __builtin_amdgcn_wmma_f32_16x16x32_f16(false, fa0.h, false, fb1.h,
                                                           (short)0, acc01, false, false);
            acc10 = __builtin_amdgcn_wmma_f32_16x16x32_f16(false, fa1.h, false, fb0.h,
                                                           (short)0, acc10, false, false);
            acc11 = __builtin_amdgcn_wmma_f32_16x16x32_f16(false, fa1.h, false, fb1.h,
                                                           (short)0, acc11, false, false);
        }
    }

    // --- epilogue: BN partial stats + strided sigma store ---
    const int nOut0 = nBase + mrow;         // column of acc*0
    const int nOut1 = nBase + 16 + mrow;    // column of acc*1
    float s0 = 0.0f, sq0 = 0.0f, s1 = 0.0f, sq1 = 0.0f;
    #pragma unroll
    for (int v = 0; v < 8; ++v) {
        float a = acc00[v], b = acc10[v], c = acc01[v], e = acc11[v];
        s0 += a + b;  sq0 += a * a + b * b;
        s1 += c + e;  sq1 += c * c + e * e;
    }
    atomicAdd(&chSum[nOut0], s0);
    atomicAdd(&chSumSq[nOut0], sq0);
    atomicAdd(&chSum[nOut1], s1);
    atomicAdd(&chSumSq[nOut1], sq1);

    // only d even, h even (M-half 0 => h = hbase, even), w even feed the output
    if (((d & 1) == 0) && (wave < 2)) {
        #pragma unroll
        for (int t = 0; t < 2; ++t) {
            #pragma unroll
            for (int v = 0; v < 8; ++v) {
                int m = 16 * t + 8 * hs + v;     // w coordinate
                if ((m & 1) == 0) {
                    float* row = sigma + (size_t)(posB + m) * 64;
                    row[nOut0] = t ? acc10[v] : acc00[v];
                    row[nOut1] = t ? acc11[v] : acc01[v];
                }
            }
        }
    }

    __syncthreads();
    if (tid < 64) {
        atomicAdd(&sumBuf[tid], chSum[tid]);
        atomicAdd(&sumsqBuf[tid], chSumSq[tid]);
    }
}

// ---------------------------------------------------------------------------
// Kernel 2: fold BN into per-channel affine: z = a*sigma + b
// ---------------------------------------------------------------------------
__global__ void pasa_finalize_stats(const float* __restrict__ sum,
                                    const float* __restrict__ sumsq,
                                    const float* __restrict__ gamma,
                                    const float* __restrict__ beta,
                                    float* __restrict__ ab) {
    int c = threadIdx.x;            // 0..63
    const float invN = 1.0f / 65536.0f;
    float mean = sum[c] * invN;
    float var  = sumsq[c] * invN - mean * mean;
    float g = (c < 54) ? gamma[c] : 0.0f;
    float b = (c < 54) ? beta[c]  : 0.0f;
    float a = g * rsqrtf(var + 1e-5f);
    ab[c]      = a;
    ab[64 + c] = b - mean * a;
}

// ---------------------------------------------------------------------------
// Kernel 3: per strided voxel: softmax over 54 BN'd channels, then 27-tap
// adaptive mix of x patches (group 0 -> out ch 0..31, group 1 -> 32..63).
// One wave per output voxel; lane handles out channels (lane, lane+32).
// ---------------------------------------------------------------------------
__global__ __launch_bounds__(256) void pasa_adaptive(
    const float* __restrict__ x, const float* __restrict__ sigma,
    const float* __restrict__ ab, float* __restrict__ out) {

    const int wid  = (blockIdx.x * 256 + threadIdx.x) >> 5;  // voxel 0..8191
    const int lane = threadIdx.x & 31;
    const int nb = wid >> 12;
    const int dc = (wid >> 8) & 15;
    const int hc = (wid >> 4) & 15;
    const int wc = wid & 15;
    const int d = dc * 2, h = hc * 2, w = wc * 2;

    const float* sig = sigma + (size_t)(nb * 32768 + d * 1024 + h * 32 + w) * 64;

    float mx = -1e30f;
    for (int c = 0; c < 54; ++c) mx = fmaxf(mx, fmaf(ab[c], sig[c], ab[64 + c]));
    float ssum = 0.0f;
    for (int c = 0; c < 54; ++c) ssum += __expf(fmaf(ab[c], sig[c], ab[64 + c]) - mx);
    const float invs = 1.0f / ssum;

    const float* x0 = x + ((size_t)nb * 64 + lane) * 32768;  // out channel = lane
    const float* x1 = x0 + (size_t)32 * 32768;               // out channel = lane+32
    float acc0 = 0.0f, acc1 = 0.0f;
    #pragma unroll
    for (int p = 0; p < 27; ++p) {
        const int di = p / 9, hi = (p / 3) % 3, wi = p % 3;
        int dd  = d + di - 1;  dd  = dd  < 0 ? 1 : (dd  > 31 ? 30 : dd);
        int hh  = h + hi - 1;  hh  = hh  < 0 ? 1 : (hh  > 31 ? 30 : hh);
        int wwv = w + wi - 1;  wwv = wwv < 0 ? 1 : (wwv > 31 ? 30 : wwv);
        const int off = dd * 1024 + hh * 32 + wwv;
        float e0 = __expf(fmaf(ab[p],      sig[p],      ab[64 + p])      - mx) * invs;
        float e1 = __expf(fmaf(ab[27 + p], sig[27 + p], ab[64 + 27 + p]) - mx) * invs;
        acc0 = fmaf(e0, x0[off], acc0);
        acc1 = fmaf(e1, x1[off], acc1);
    }
    const size_t ob = (size_t)nb * 64 * 4096 + dc * 256 + hc * 16 + wc;
    out[ob + (size_t)lane * 4096]        = acc0;
    out[ob + (size_t)(lane + 32) * 4096] = acc1;
}

// ---------------------------------------------------------------------------
// Workspace layout (bytes):
//   [0, 16777216)            sigma f32 (65536 pos x 64 ch), strided voxels only
//   [16777216, 16998400)     packed B f16 (27*64*64)
//   [16998400, 16998656)     per-channel sum
//   [16998656, 16998912)     per-channel sumsq
//   [16998912, 16999424)     ab affine (a[64], b[64])
// ---------------------------------------------------------------------------
extern "C" void kernel_launch(void* const* d_in, const int* in_sizes, int n_in,
                              void* d_out, int out_size, void* d_ws, size_t ws_size,
                              hipStream_t stream) {
    const float* x      = (const float*)d_in[0];
    const float* conv_w = (const float*)d_in[1];
    const float* gamma  = (const float*)d_in[2];
    const float* beta   = (const float*)d_in[3];
    float* out = (float*)d_out;

    char* ws = (char*)d_ws;
    float*    sigma    = (float*)ws;
    _Float16* Bp       = (_Float16*)(ws + 16777216);
    float*    sumBuf   = (float*)(ws + 16998400);
    float*    sumsqBuf = (float*)(ws + 16998656);
    float*    ab       = (float*)(ws + 16998912);

    hipMemsetAsync(sumBuf, 0, 512, stream);   // zero sum + sumsq
    pasa_pack_b<<<432, 256, 0, stream>>>(conv_w, Bp);
    pasa_conv_wmma<<<1024, 128, 0, stream>>>(x, Bp, sigma, sumBuf, sumsqBuf);
    pasa_finalize_stats<<<1, 64, 0, stream>>>(sumBuf, sumsqBuf, gamma, beta, ab);
    pasa_adaptive<<<1024, 256, 0, stream>>>(x, sigma, ab, out);
}
